// UDUModel_76888504533074
// MI455X (gfx1250) — matrix operand order repaired and checked by reading, not compiled, not run on hardware
//
#include <hip/hip_runtime.h>

// ---------------------------------------------------------------------------
// UDU point-cloud model forward for gfx1250 (CDNA5, wave32, WMMA).
// Dense layers -> v_wmma_f32_16x16x32_f16 with double-buffered LDS tiles
// (batched branch-free clamped global loads, packed f16 LDS, ds_load_b128
// fragment assembly), fused bias / eval-BN / ReLU / residual epilogue.
// FPS / kNN / attention use GLOBAL_LOAD_ASYNC_TO_LDS_B32 for their
// global->LDS staging (ASYNCcnt path).
// ---------------------------------------------------------------------------

#define BN_INV 0.99999500003749968f   // 1/sqrt(1 + 1e-5)

typedef __attribute__((ext_vector_type(16))) _Float16 f16x16;
typedef __attribute__((ext_vector_type(8)))  _Float16 f16x8;
typedef __attribute__((ext_vector_type(2)))  _Float16 f16x2;
typedef __attribute__((ext_vector_type(8)))  float    f32x8;

#define TPAD 8   // f16 row pad: keeps 16B alignment (stride 80B), staggers banks

// ---- CDNA5 async global->LDS copy (tracked on ASYNCcnt) ----
__device__ __forceinline__ void async_copy_f32(float* lds_dst, const float* gsrc)
{
    unsigned lo = (unsigned)(size_t)lds_dst;   // low 32 bits of generic ptr == LDS offset
    asm volatile("global_load_async_to_lds_b32 %0, %1, off"
                 :: "v"(lo), "v"(gsrc) : "memory");
}
__device__ __forceinline__ void async_wait0()
{
    asm volatile("s_wait_asynccnt 0" ::: "memory");
}

// ======================= WMMA GEMM =======================
// Y[M,N] = X[M,K] @ W[N,K]^T  (+bias[n]) (optional BN: *g*BN_INV+be) (opt relu)
// (optional residual add, residual layout == Y layout)
// Block = 128 threads = 4 waves; block tile 64x16; each wave owns one 16x16.
// Double-buffered LDS tiles: stage chunk c+1 while WMMA consumes chunk c.
__global__ void gemm_wmma_kernel(const float* __restrict__ X,
                                 const float* __restrict__ W,
                                 const float* __restrict__ bias,
                                 const float* __restrict__ gamma,
                                 const float* __restrict__ beta,
                                 const float* __restrict__ resid,
                                 float* __restrict__ Y,
                                 int M, int N, int K, int relu)
{
    __shared__ __align__(16) _Float16 As[2][64][32 + TPAD];
    __shared__ __align__(16) _Float16 Bs[2][16][32 + TPAD];

    const int tid  = threadIdx.x;
    const int lane = tid & 31;
    const int wid  = tid >> 5;
    const int hf   = lane >> 4;     // half-wave (0/1)
    const int lr   = lane & 15;
    const int mBase = blockIdx.y * 64;
    const int m0    = mBase + wid * 16;
    const int n0    = blockIdx.x * 16;

    f32x8 acc = {0.f,0.f,0.f,0.f,0.f,0.f,0.f,0.f};

    // Stage one 64x32 A tile + 16x32 B tile as packed f16.
    // Phase 1: issue ALL global loads (clamped addresses, independent) so the
    // compiler batches them behind a single s_wait_loadcnt.
    // Phase 2: convert + pack + ds_store.
    auto stage = [&](int p, int k0) {
        float va[16];
        float vb[4];
#pragma unroll
        for (int j = 0; j < 8; ++j) {
            int linear = j * 256 + tid * 2;         // 2048 elems / 128 thr
            int row = linear >> 5, kk = linear & 31;
            int gr  = mBase + row;
            int cr  = gr < M ? gr : M - 1;
            int gk0 = k0 + kk, gk1 = gk0 + 1;
            const float* xr = X + (size_t)cr * K;
            va[2*j]   = xr[gk0 < K ? gk0 : K - 1];
            va[2*j+1] = xr[gk1 < K ? gk1 : K - 1];
        }
#pragma unroll
        for (int j = 0; j < 2; ++j) {
            int linear = j * 256 + tid * 2;         // 512 elems / 128 thr
            int col = linear >> 5, kk = linear & 31;
            int gc  = n0 + col;
            int cc  = gc < N ? gc : N - 1;
            int gk0 = k0 + kk, gk1 = gk0 + 1;
            const float* wr = W + (size_t)cc * K;
            vb[2*j]   = wr[gk0 < K ? gk0 : K - 1];
            vb[2*j+1] = wr[gk1 < K ? gk1 : K - 1];
        }
#pragma unroll
        for (int j = 0; j < 8; ++j) {
            int linear = j * 256 + tid * 2;
            int row = linear >> 5, kk = linear & 31;
            int gr  = mBase + row;
            int gk0 = k0 + kk, gk1 = gk0 + 1;
            f16x2 pk;
            pk[0] = (gr < M && gk0 < K) ? (_Float16)va[2*j]   : (_Float16)0.f;
            pk[1] = (gr < M && gk1 < K) ? (_Float16)va[2*j+1] : (_Float16)0.f;
            *(f16x2*)&As[p][row][kk] = pk;
        }
#pragma unroll
        for (int j = 0; j < 2; ++j) {
            int linear = j * 256 + tid * 2;
            int col = linear >> 5, kk = linear & 31;
            int gc  = n0 + col;
            int gk0 = k0 + kk, gk1 = gk0 + 1;
            f16x2 pk;
            pk[0] = (gc < N && gk0 < K) ? (_Float16)vb[2*j]   : (_Float16)0.f;
            pk[1] = (gc < N && gk1 < K) ? (_Float16)vb[2*j+1] : (_Float16)0.f;
            *(f16x2*)&Bs[p][col][kk] = pk;
        }
    };

    const int nChunks = (K + 31) >> 5;
    stage(0, 0);
    for (int c = 0; c < nChunks; ++c) {
        __syncthreads();   // stage(c) visible; compute(c-1) done -> buffer (c+1)&1 free
        if (c + 1 < nChunks) stage((c + 1) & 1, (c + 1) << 5);

        const int p = c & 1;
        // ---- fragment assembly straight from LDS (aligned 16B vector loads)
        // A (ISA 7.12.2): elem e<8 -> K = 8*hf + e ; e>=8 -> K = 16 + 8*hf + (e-8)
        f16x8 a0 = *(const f16x8*)&As[p][wid * 16 + lr][hf * 8];
        f16x8 a1 = *(const f16x8*)&As[p][wid * 16 + lr][16 + hf * 8];
        // B: same as A: elem e<8 -> K = 8*hf + e ; e>=8 -> K = 16 + 8*hf + (e-8)
        f16x8 b0 = *(const f16x8*)&Bs[p][lr][hf * 8];
        f16x8 b1 = *(const f16x8*)&Bs[p][lr][16 + hf * 8];
        f16x16 a = __builtin_shufflevector(a0, a1, 0,1,2,3,4,5,6,7,8,9,10,11,12,13,14,15);
        f16x16 b = __builtin_shufflevector(b0, b1, 0,1,2,3,4,5,6,7,8,9,10,11,12,13,14,15);

        acc = __builtin_amdgcn_wmma_f32_16x16x32_f16(false, a, false, b,
                                                     (short)0, acc, false, false);
    }

    const int bcol = n0 + lr;
    if (bcol < N) {
        float bs = bias ? bias[bcol] : 0.f;
        float sc = 1.f, sh = 0.f;
        if (gamma) { sc = gamma[bcol] * BN_INV; sh = beta[bcol]; }
#pragma unroll
        for (int v = 0; v < 8; ++v) {
            int row = m0 + (hf << 3) + v;   // C/D layout: m = v + 8*half
            if (row < M) {
                float t = acc[v] + bs;
                if (gamma) t = t * sc + sh;
                if (relu)  t = fmaxf(t, 0.f);
                if (resid) t += resid[(size_t)row * N + bcol];
                Y[(size_t)row * N + bcol] = t;
            }
        }
    }
}

// ======================= Farthest point sampling =======================
// One block per batch. dist[] lives in LDS; sequential npoint iterations of
// (update min-dist, block argmax). Deterministic start at index 0,
// ties -> lowest index (matches jnp.argmax).
__global__ void fps_kernel(const float* __restrict__ X, int* __restrict__ fidx,
                           int N, int D, int np)
{
    extern __shared__ float sm[];
    float* dist = sm;                       // N
    float* pv   = dist + N;                 // D
    float* rv   = pv + D;                   // BLK
    int*   ri   = (int*)(rv + blockDim.x);  // BLK
    const int b = blockIdx.x, tid = threadIdx.x, BLK = blockDim.x;

    for (int i = tid; i < N; i += BLK) dist[i] = 1e10f;
    if (tid == 0) fidx[(size_t)b * np] = 0;
    int last = 0;
    __syncthreads();

    for (int s = 1; s < np; ++s) {
        // async DMA the pivot point into LDS
        for (int d = tid; d < D; d += BLK)
            async_copy_f32(&pv[d], &X[((size_t)b * N + last) * D + d]);
        async_wait0();
        __syncthreads();
        float best = -1.f; int bi = 0;
        for (int i = tid; i < N; i += BLK) {
            const float* xr = X + ((size_t)b * N + i) * D;
            float acc = 0.f;
            for (int d = 0; d < D; ++d) { float df = xr[d] - pv[d]; acc += df * df; }
            float nd = fminf(dist[i], acc);
            dist[i] = nd;
            if (nd > best) { best = nd; bi = i; }
        }
        rv[tid] = best; ri[tid] = bi;
        __syncthreads();
        for (int off = BLK >> 1; off > 0; off >>= 1) {
            if (tid < off) {
                if (rv[tid+off] > rv[tid] ||
                    (rv[tid+off] == rv[tid] && ri[tid+off] < ri[tid])) {
                    rv[tid] = rv[tid+off]; ri[tid] = ri[tid+off];
                }
            }
            __syncthreads();
        }
        last = ri[0];
        if (tid == 0) fidx[(size_t)b * np + s] = last;
        __syncthreads();
    }
}

// ======================= kNN (k=32, squared distance) =======================
// One block per query point; distances in LDS; 32 rounds of block argmin
// (ties -> lowest index, matching stable top_k).
__global__ void knn_kernel(const float* __restrict__ Qp, const float* __restrict__ Ref,
                           int* __restrict__ nidx, int M, int N, int D, int Kn)
{
    extern __shared__ float sm[];
    float* qv   = sm;                        // D
    float* dist = qv + D;                    // N
    float* rv   = dist + N;                  // BLK
    int*   ri   = (int*)(rv + blockDim.x);   // BLK
    const int bid = blockIdx.x;
    const int m = bid % M, b = bid / M;
    const int tid = threadIdx.x, BLK = blockDim.x;

    const float* qrow = Qp + ((size_t)b * M + m) * D;
    for (int d = tid; d < D; d += BLK)
        async_copy_f32(&qv[d], &qrow[d]);
    async_wait0();
    __syncthreads();
    for (int i = tid; i < N; i += BLK) {
        const float* rr = Ref + ((size_t)b * N + i) * D;
        float acc = 0.f;
        for (int d = 0; d < D; ++d) { float df = qv[d] - rr[d]; acc += df * df; }
        dist[i] = acc;
    }
    __syncthreads();
    for (int s = 0; s < Kn; ++s) {
        float best = 3.4e38f; int bi = 0x7fffffff;
        for (int i = tid; i < N; i += BLK) {
            float v = dist[i];
            if (v < best) { best = v; bi = i; }
        }
        rv[tid] = best; ri[tid] = bi;
        __syncthreads();
        for (int off = BLK >> 1; off > 0; off >>= 1) {
            if (tid < off) {
                if (rv[tid+off] < rv[tid] ||
                    (rv[tid+off] == rv[tid] && ri[tid+off] < ri[tid])) {
                    rv[tid] = rv[tid+off]; ri[tid] = ri[tid+off];
                }
            }
            __syncthreads();
        }
        int sel = ri[0];
        if (tid == 0) {
            nidx[((size_t)b * M + m) * Kn + s] = sel;
            dist[sel] = 3.4e38f;
        }
        __syncthreads();
    }
}

// gather: out[b,m,:] = X[b, idx[b,m], :]
__global__ void gather_kernel(const float* __restrict__ X, const int* __restrict__ idx,
                              float* __restrict__ out, int B, int N, int M, int D)
{
    int i = blockIdx.x * blockDim.x + threadIdx.x;
    int total = B * M * D;
    if (i >= total) return;
    int d = i % D, m = (i / D) % M, b = i / (D * M);
    int n = idx[(size_t)b * M + m];
    out[i] = X[((size_t)b * N + n) * D + d];
}

// group + max over k neighbors: out[b,m,c] with c<[Cp: points | D: rel-xyz]
__global__ void groupmax_kernel(const float* __restrict__ P, const float* __restrict__ xyz,
                                const float* __restrict__ newxyz, const int* __restrict__ nidx,
                                float* __restrict__ out, int B, int N, int M,
                                int Cp, int D, int Kn)
{
    int C = Cp + D;
    int i = blockIdx.x * blockDim.x + threadIdx.x;
    int total = B * M * C;
    if (i >= total) return;
    int c = i % C, m = (i / C) % M, b = i / (C * M);
    const int* nb = nidx + ((size_t)b * M + m) * Kn;
    float best = -3.4e38f;
    if (c < Cp) {
        for (int j = 0; j < Kn; ++j)
            best = fmaxf(best, P[((size_t)b * N + nb[j]) * Cp + c]);
    } else {
        int d = c - Cp;
        float ctr = newxyz[((size_t)b * M + m) * D + d];
        for (int j = 0; j < Kn; ++j)
            best = fmaxf(best, xyz[((size_t)b * N + nb[j]) * D + d] - ctr);
    }
    out[i] = best;
}

// global max-pool over N points, write into fv slice
__global__ void maxpool_kernel(const float* __restrict__ X, float* __restrict__ out,
                               int B, int N, int C, int outStride, int outOff)
{
    int i = blockIdx.x * blockDim.x + threadIdx.x;
    if (i >= B * C) return;
    int b = i / C, c = i % C;
    float best = -3.4e38f;
    for (int n = 0; n < N; ++n)
        best = fmaxf(best, X[((size_t)b * N + n) * C + c]);
    out[(size_t)b * outStride + outOff + c] = best;
}

// concat along channels: out[r] = [A(Ca) | Bs(Cb)]
__global__ void concat2_kernel(const float* __restrict__ A, int Ca,
                               const float* __restrict__ Bs, int Cb,
                               float* __restrict__ out, int Mrows)
{
    int C = Ca + Cb;
    int i = blockIdx.x * blockDim.x + threadIdx.x;
    if (i >= Mrows * C) return;
    int r = i / C, c = i % C;
    out[i] = (c < Ca) ? A[(size_t)r * Ca + c] : Bs[(size_t)r * Cb + (c - Ca)];
}

// build folding input F[b,m,0:261] = [feat(256) | seed(2) | pf(3)], also emit pf
__global__ void build_fold_kernel(const float* __restrict__ feat,
                                  const float* __restrict__ coarse,
                                  float* __restrict__ Fo, float* __restrict__ pf,
                                  int B, int nc)
{
    int nd = nc * 4, C = 261;
    int i = blockIdx.x * blockDim.x + threadIdx.x;
    int total = B * nd * C;
    if (i >= total) return;
    int c = i % C, m = (i / C) % nd, b = i / (C * nd);
    float v;
    if (c < 256)       v = feat[(size_t)b * 256 + c];
    else if (c == 256) v = (m & 1) ? 0.05f : -0.05f;      // fast axis
    else if (c == 257) v = (m & 2) ? 0.05f : -0.05f;      // slow axis
    else {
        int d = c - 258;
        v = coarse[((size_t)b * nc + (m >> 2)) * 3 + d];
        pf[((size_t)b * nd + m) * 3 + d] = v;
    }
    Fo[i] = v;
}

// ======================= attention softmax(QK^T/sqrt(d)) V =======================
// One block per (b, head, query). Scores kept in LDS.
__global__ void attn_kernel(const float* __restrict__ Q, const float* __restrict__ Km,
                            const float* __restrict__ Vm, float* __restrict__ O,
                            int B, int Lq, int Lk, int H, int dh)
{
    extern __shared__ float sm[];
    float* qs  = sm;                 // dh
    float* sc  = qs + dh;            // Lk
    float* red = sc + Lk;            // BLK
    const int bid = blockIdx.x;
    const int q = bid % Lq, h = (bid / Lq) % H, b = bid / (Lq * H);
    const int E = H * dh;
    const int tid = threadIdx.x, BLK = blockDim.x;

    const float* qrow = Q + ((size_t)(b * Lq + q)) * E + h * dh;
    for (int d = tid; d < dh; d += BLK)
        async_copy_f32(&qs[d], &qrow[d]);
    async_wait0();
    __syncthreads();

    const float scale = rsqrtf((float)dh);
    float lmax = -3.4e38f;
    for (int j = tid; j < Lk; j += BLK) {
        const float* kr = Km + ((size_t)(b * Lk + j)) * E + h * dh;
        float acc = 0.f;
        for (int d = 0; d < dh; ++d) acc += qs[d] * kr[d];
        acc *= scale;
        sc[j] = acc;
        lmax = fmaxf(lmax, acc);
    }
    red[tid] = lmax; __syncthreads();
    for (int off = BLK >> 1; off > 0; off >>= 1) {
        if (tid < off) red[tid] = fmaxf(red[tid], red[tid + off]);
        __syncthreads();
    }
    float mx = red[0]; __syncthreads();

    float lsum = 0.f;
    for (int j = tid; j < Lk; j += BLK) { float e = expf(sc[j] - mx); sc[j] = e; lsum += e; }
    red[tid] = lsum; __syncthreads();
    for (int off = BLK >> 1; off > 0; off >>= 1) {
        if (tid < off) red[tid] += red[tid + off];
        __syncthreads();
    }
    float inv = 1.f / red[0]; __syncthreads();

    float oacc[32];
    for (int d = 0; d < 32; ++d) oacc[d] = 0.f;
    for (int j = tid; j < Lk; j += BLK) {
        float p = sc[j] * inv;
        const float* vr = Vm + ((size_t)(b * Lk + j)) * E + h * dh;
        for (int d = 0; d < dh; ++d) oacc[d] += p * vr[d];
    }
    for (int d = 0; d < dh; ++d) {
        red[tid] = oacc[d]; __syncthreads();
        for (int off = BLK >> 1; off > 0; off >>= 1) {
            if (tid < off) red[tid] += red[tid + off];
            __syncthreads();
        }
        if (tid == 0) O[((size_t)(b * Lq + q)) * E + h * dh + d] = red[0];
        __syncthreads();
    }
}

// ======================= host orchestration =======================

static void gemm(hipStream_t st, const float* X, const float* W, const float* bias,
                 const float* gamma, const float* beta, const float* resid,
                 float* Y, int M, int N, int K, int relu)
{
    dim3 g((N + 15) / 16, (M + 63) / 64);
    gemm_wmma_kernel<<<g, dim3(128, 1, 1), 0, st>>>(X, W, bias, gamma, beta, resid,
                                                    Y, M, N, K, relu);
}

static inline int nblk(long long total, int bs) { return (int)((total + bs - 1) / bs); }

struct PD { const float *w1,*b1,*g1,*be1,*w2,*b2,*w3,*b3,*g2,*be2,*w4,*b4; };
struct PU { const float *l1,*lb1,*l2,*lb2,*l3,*lb3,*c1,*cb1,*g1,*be1,*c2,*cb2,*g2,*be2,*c3,*cb3; };

extern "C" void kernel_launch(void* const* d_in, const int* in_sizes, int n_in,
                              void* d_out, int out_size, void* d_ws, size_t ws_size,
                              hipStream_t stream)
{
    (void)in_sizes; (void)n_in; (void)out_size; (void)ws_size;
    auto F = [&](int i) { return (const float*)d_in[i]; };
    // Depth-first, insertion-order flattening of setup_inputs():
    // 0:x  e1b1:1..12  e1b2:13..24  e1b3:25..36  dec1:37..52  dec2:53..68
    // e2b1:69..80  e2b2:81..92  e2b3:93..104  fl1w:105 fl1b:106 fl2w:107 fl2b:108
    // attn:109..116
    auto pd = [&](int o) {
        PD p = { F(o),F(o+1),F(o+2),F(o+3),F(o+4),F(o+5),F(o+6),F(o+7),F(o+8),F(o+9),F(o+10),F(o+11) };
        return p;
    };
    auto pu = [&](int o) {
        PU p = { F(o),F(o+1),F(o+2),F(o+3),F(o+4),F(o+5),F(o+6),F(o+7),
                 F(o+8),F(o+9),F(o+10),F(o+11),F(o+12),F(o+13),F(o+14),F(o+15) };
        return p;
    };
    const float* x = F(0);
    PD e1b1 = pd(1), e1b2 = pd(13), e1b3 = pd(25);
    PU dec1 = pu(37), dec2 = pu(53);
    PD e2b1 = pd(69), e2b2 = pd(81), e2b3 = pd(93);
    const float *fl1w = F(105), *fl1b = F(106), *fl2w = F(107), *fl2b = F(108);
    const float *wq = F(109), *bq = F(110), *wk = F(111), *bk = F(112),
                *wv = F(113), *bv = F(114), *wo = F(115), *bo = F(116);

    // ---- workspace (float elements, bump allocated) ----
    float* ws = (float*)d_ws;
    size_t off = 0;
    auto alloc = [&](size_t n) { float* p = ws + off; off += n; return p; };
    float* bufA = alloc(8388608);   // up to [16384,512]
    float* bufB = alloc(8388608);   // up to [16384,512]
    float* bufC = alloc(4276224);   // up to [16384,261]
    float* X1   = alloc(524288);    // [8192,64]
    float* X2   = alloc(524288);    // [4096,128]
    float* X3   = alloc(524288);    // [2048,256]
    float* YB   = alloc(2097152);   // [8192,256]
    float* YC   = alloc(524288);    // [4096,128]
    float* Y0   = alloc(49152);     // [16384,3]
    float* C1p  = alloc(12288);     // [4096,3]
    float* PFB  = alloc(49152);     // pf, up to [16384,3]
    float* FV   = alloc(1792);      // [4,448]
    float* F1   = alloc(1024);      // [4,256]
    float* F2   = alloc(1024);      // [4,256]
    float* COAR = alloc(3072);      // [4,768]
    float* NEWX = alloc(1048576);   // up to [4096,256]
    int*   NIDX = (int*)alloc(262144);
    int*   FIDX = (int*)alloc(8192);

    const int BLK = 256;
    const int Bt = 4;

    // =========== stage A: x1 = pcdown(e1b1, x, reduction=False) ===========
    gemm(stream, x, e1b1.w1, e1b1.b1, e1b1.g1, e1b1.be1, nullptr, bufA, 8192, 32, 3, 1);
    gemm(stream, bufA, e1b1.w2, e1b1.b2, nullptr, nullptr, nullptr, bufB, 8192, 64, 32, 0);
    concat2_kernel<<<nblk(8192LL*67, BLK), BLK, 0, stream>>>(bufB, 64, x, 3, bufC, 8192);
    gemm(stream, bufC, e1b1.w3, e1b1.b3, e1b1.g2, e1b1.be2, nullptr, bufA, 8192, 64, 67, 1);
    gemm(stream, bufA, e1b1.w4, e1b1.b4, nullptr, nullptr, nullptr, X1, 8192, 64, 64, 0);

    // =========== stage B: x2 = pcdown(e1b2, x1) : 2048 -> 1024 ===========
    gemm(stream, X1, e1b2.w1, e1b2.b1, e1b2.g1, e1b2.be1, nullptr, bufA, 8192, 64, 64, 1);
    gemm(stream, bufA, e1b2.w2, e1b2.b2, nullptr, nullptr, nullptr, bufB, 8192, 128, 64, 0);
    {
        int N = 2048, M = 1024, D = 64;
        fps_kernel<<<Bt, BLK, (size_t)(N + D + 2*BLK)*4, stream>>>(X1, FIDX, N, D, M);
        gather_kernel<<<nblk((long long)Bt*M*D, BLK), BLK, 0, stream>>>(X1, FIDX, NEWX, Bt, N, M, D);
        knn_kernel<<<Bt*M, BLK, (size_t)(D + N + 2*BLK)*4, stream>>>(NEWX, X1, NIDX, M, N, D, 32);
        groupmax_kernel<<<nblk((long long)Bt*M*(128+D), BLK), BLK, 0, stream>>>(
            bufB, X1, NEWX, NIDX, bufC, Bt, N, M, 128, D, 32);
    }
    gemm(stream, bufC, e1b2.w3, e1b2.b3, e1b2.g2, e1b2.be2, nullptr, bufA, 4096, 128, 192, 1);
    gemm(stream, bufA, e1b2.w4, e1b2.b4, nullptr, nullptr, nullptr, X2, 4096, 128, 128, 0);

    // =========== stage C: x3 = pcdown(e1b3, x2) : 1024 -> 512 ===========
    gemm(stream, X2, e1b3.w1, e1b3.b1, e1b3.g1, e1b3.be1, nullptr, bufA, 4096, 128, 128, 1);
    gemm(stream, bufA, e1b3.w2, e1b3.b2, nullptr, nullptr, nullptr, bufB, 4096, 256, 128, 0);
    {
        int N = 1024, M = 512, D = 128;
        fps_kernel<<<Bt, BLK, (size_t)(N + D + 2*BLK)*4, stream>>>(X2, FIDX, N, D, M);
        gather_kernel<<<nblk((long long)Bt*M*D, BLK), BLK, 0, stream>>>(X2, FIDX, NEWX, Bt, N, M, D);
        knn_kernel<<<Bt*M, BLK, (size_t)(D + N + 2*BLK)*4, stream>>>(NEWX, X2, NIDX, M, N, D, 32);
        groupmax_kernel<<<nblk((long long)Bt*M*(256+D), BLK), BLK, 0, stream>>>(
            bufB, X2, NEWX, NIDX, bufC, Bt, N, M, 256, D, 32);
    }
    gemm(stream, bufC, e1b3.w3, e1b3.b3, e1b3.g2, e1b3.be2, nullptr, bufA, 2048, 256, 384, 1);
    gemm(stream, bufA, e1b3.w4, e1b3.b4, nullptr, nullptr, nullptr, X3, 2048, 256, 256, 0);

    // =========== stage D: fv + f1/f2 ===========
    maxpool_kernel<<<nblk(Bt*64, BLK), BLK, 0, stream>>>(X1, FV, Bt, 2048, 64, 448, 0);
    maxpool_kernel<<<nblk(Bt*128, BLK), BLK, 0, stream>>>(X2, FV, Bt, 1024, 128, 448, 64);
    maxpool_kernel<<<nblk(Bt*256, BLK), BLK, 0, stream>>>(X3, FV, Bt, 512, 256, 448, 192);
    gemm(stream, FV, fl1w, fl1b, nullptr, nullptr, nullptr, F1, 4, 256, 448, 0);
    gemm(stream, FV, fl2w, fl2b, nullptr, nullptr, nullptr, F2, 4, 256, 448, 0);

    // =========== stage E: c1 = pcup(dec1, f1, 256) ===========
    gemm(stream, F1, dec1.l1, dec1.lb1, nullptr, nullptr, nullptr, bufA, 4, 1024, 256, 1);
    gemm(stream, bufA, dec1.l2, dec1.lb2, nullptr, nullptr, nullptr, bufB, 4, 1024, 1024, 1);
    gemm(stream, bufB, dec1.l3, dec1.lb3, nullptr, nullptr, nullptr, COAR, 4, 768, 1024, 0);
    build_fold_kernel<<<nblk(4LL*1024*261, BLK), BLK, 0, stream>>>(F1, COAR, bufC, PFB, Bt, 256);
    gemm(stream, bufC, dec1.c1, dec1.cb1, dec1.g1, dec1.be1, nullptr, bufA, 4096, 512, 261, 1);
    gemm(stream, bufA, dec1.c2, dec1.cb2, dec1.g2, dec1.be2, nullptr, bufB, 4096, 512, 512, 1);
    gemm(stream, bufB, dec1.c3, dec1.cb3, nullptr, nullptr, PFB, C1p, 4096, 3, 512, 0);

    // =========== stage F: y0 = pcup(dec2, f2, 1024, coarse=c1) ===========
    build_fold_kernel<<<nblk(4LL*4096*261, BLK), BLK, 0, stream>>>(F2, C1p, bufC, PFB, Bt, 1024);
    gemm(stream, bufC, dec2.c1, dec2.cb1, dec2.g1, dec2.be1, nullptr, bufA, 16384, 512, 261, 1);
    gemm(stream, bufA, dec2.c2, dec2.cb2, dec2.g2, dec2.be2, nullptr, bufB, 16384, 512, 512, 1);
    gemm(stream, bufB, dec2.c3, dec2.cb3, nullptr, nullptr, PFB, Y0, 16384, 3, 512, 0);

    // =========== stage G: pcdown(e2b1, y0) : 4096 -> 2048 ===========
    gemm(stream, Y0, e2b1.w1, e2b1.b1, e2b1.g1, e2b1.be1, nullptr, bufA, 16384, 128, 3, 1);
    gemm(stream, bufA, e2b1.w2, e2b1.b2, nullptr, nullptr, nullptr, bufB, 16384, 256, 128, 0);
    {
        int N = 4096, M = 2048, D = 3;
        fps_kernel<<<Bt, BLK, (size_t)(N + D + 2*BLK)*4, stream>>>(Y0, FIDX, N, D, M);
        gather_kernel<<<nblk((long long)Bt*M*D, BLK), BLK, 0, stream>>>(Y0, FIDX, NEWX, Bt, N, M, D);
        knn_kernel<<<Bt*M, BLK, (size_t)(D + N + 2*BLK)*4, stream>>>(NEWX, Y0, NIDX, M, N, D, 32);
        groupmax_kernel<<<nblk((long long)Bt*M*(256+D), BLK), BLK, 0, stream>>>(
            bufB, Y0, NEWX, NIDX, bufC, Bt, N, M, 256, D, 32);
    }
    gemm(stream, bufC, e2b1.w3, e2b1.b3, e2b1.g2, e2b1.be2, nullptr, bufA, 8192, 256, 259, 1);
    gemm(stream, bufA, e2b1.w4, e2b1.b4, nullptr, nullptr, nullptr, YB, 8192, 256, 256, 0);

    // =========== stage H: pcdown(e2b2, y) : 2048 -> 1024 ===========
    gemm(stream, YB, e2b2.w1, e2b2.b1, e2b2.g1, e2b2.be1, nullptr, bufA, 8192, 128, 256, 1);
    gemm(stream, bufA, e2b2.w2, e2b2.b2, nullptr, nullptr, nullptr, bufB, 8192, 128, 128, 0);
    {
        int N = 2048, M = 1024, D = 256;
        fps_kernel<<<Bt, BLK, (size_t)(N + D + 2*BLK)*4, stream>>>(YB, FIDX, N, D, M);
        gather_kernel<<<nblk((long long)Bt*M*D, BLK), BLK, 0, stream>>>(YB, FIDX, NEWX, Bt, N, M, D);
        knn_kernel<<<Bt*M, BLK, (size_t)(D + N + 2*BLK)*4, stream>>>(NEWX, YB, NIDX, M, N, D, 32);
        groupmax_kernel<<<nblk((long long)Bt*M*(128+D), BLK), BLK, 0, stream>>>(
            bufB, YB, NEWX, NIDX, bufC, Bt, N, M, 128, D, 32);
    }
    gemm(stream, bufC, e2b2.w3, e2b2.b3, e2b2.g2, e2b2.be2, nullptr, bufA, 4096, 128, 384, 1);
    gemm(stream, bufA, e2b2.w4, e2b2.b4, nullptr, nullptr, nullptr, YC, 4096, 128, 128, 0);

    // =========== stage I: y = y + mha(y, c1) ===========
    float* Qb = bufA;
    float* Kb = bufA + 524288;
    float* Vb = bufA + 1048576;
    float* Ob = bufA + 1572864;
    gemm(stream, YC, wq, bq, nullptr, nullptr, nullptr, Qb, 4096, 128, 128, 0);
    gemm(stream, C1p, wk, bk, nullptr, nullptr, nullptr, Kb, 4096, 128, 3, 0);
    gemm(stream, C1p, wv, bv, nullptr, nullptr, nullptr, Vb, 4096, 128, 3, 0);
    attn_kernel<<<Bt*4*1024, 128, (size_t)(32 + 1024 + 128)*4, stream>>>(
        Qb, Kb, Vb, Ob, Bt, 1024, 1024, 4, 32);
    gemm(stream, Ob, wo, bo, nullptr, nullptr, YC, bufB, 4096, 128, 128, 0);  // bufB = yD

    // =========== stage J: pcdown(e2b3, yD, reduction=False) -> d_out ===========
    gemm(stream, bufB, e2b3.w1, e2b3.b1, e2b3.g1, e2b3.be1, nullptr, bufA, 4096, 64, 128, 1);
    gemm(stream, bufA, e2b3.w2, e2b3.b2, nullptr, nullptr, nullptr, bufC, 4096, 3, 64, 0);
    concat2_kernel<<<nblk(4096LL*131, BLK), BLK, 0, stream>>>(bufC, 3, bufB, 128, NEWX, 4096);
    gemm(stream, NEWX, e2b3.w3, e2b3.b3, e2b3.g2, e2b3.be2, nullptr, bufA, 4096, 3, 131, 1);
    gemm(stream, bufA, e2b3.w4, e2b3.b4, nullptr, nullptr, nullptr, (float*)d_out, 4096, 3, 3, 0);
}
